// MyLSTMEncoder_41266045780002
// MI455X (gfx1250) — matrix operand order repaired and checked by reading, not compiled, not run on hardware
//
#include <hip/hip_runtime.h>

#define Tn   1024
#define Bn   64
#define Hn   512
#define HS   16     // hidden units per workgroup
#define NRr  64     // gate rows per workgroup (4 gates * HS)
#define NWG  32
#define LDK  512    // k-stride (halves) of x/h sequence buffers

typedef __attribute__((ext_vector_type(16))) _Float16 v16h;
typedef __attribute__((ext_vector_type(8)))  _Float16 v8h;
typedef __attribute__((ext_vector_type(8)))  float    v8f;

#define WMMA_F16(a, b, c) \
  __builtin_amdgcn_wmma_f32_16x16x32_f16(false, (a), false, (b), (short)0, (c), false, false)

// ---- A fragment (16x32 f16, row-major source, ld = LDK halves) -------------
__device__ __forceinline__ v16h load_a_frag(const _Float16* __restrict__ p0) {
  int lane = threadIdx.x & 31;
  int r    = lane & 15;
  int hi   = lane >> 4;
  const _Float16* p = p0 + (size_t)r * LDK + hi * 8;
  v8h lo = *(const v8h*)p;
  v8h hh = *(const v8h*)(p + 16);
  v16h a;
#pragma unroll
  for (int i = 0; i < 8; ++i) { a[i] = lo[i]; a[8 + i] = hh[i]; }
  return a;
}

// ---- B fragment (32x16) from pre-swizzled f16 WT[k][row], lane = N ---------
__device__ __forceinline__ v16h load_b_frag16(const _Float16* __restrict__ WT,
                                              int row0, int k, int ld) {
  int lane = threadIdx.x & 31;
  const _Float16* p = WT + (size_t)(k + (lane >> 4) * 8) * ld + row0 + (lane & 15);
  v16h b;
#pragma unroll
  for (int i = 0; i < 8; ++i) { b[i] = p[(size_t)i * ld]; b[8 + i] = p[(size_t)(16 + i) * ld]; }
  return b;
}

__device__ __forceinline__ void grid_barrier(unsigned* bar, unsigned target) {
  __syncthreads();
  if (threadIdx.x == 0) {
    __threadfence();
    atomicAdd(bar, 1u);
    while (__hip_atomic_load(bar, __ATOMIC_RELAXED, __HIP_MEMORY_SCOPE_AGENT) < target) {
      __builtin_amdgcn_s_sleep(2);
    }
    __threadfence();
  }
  __syncthreads();
}

__device__ __forceinline__ float sigmoidf_(float x) {
  return 1.0f / (1.0f + __expf(-x));
}

// ---------------------------------------------------------------------------
// One LSTM layer, persistent kernel. 32 WGs x 512 threads (16 waves).
// Wave = g*4 + kq: gate g (16 cols), K-quarter kq (128 of 512 k's).
// Each wave keeps its unique 8 B-fragments (64 VGPRs) resident and produces
// K-partial gate tiles for all 4 batch tiles; partials reduced via LDS.
// hseq: (T+1, 64, 512) f16; slice 0 zeros (h_{-1}); h_t at slice t+1.
// xin : same layout, x_t at slice t+1 (prev layer output). Unused if LAYER0.
// ---------------------------------------------------------------------------
template <bool LAYER0>
__global__ __launch_bounds__(512) void lstm_layer(
    const _Float16* __restrict__ xin, const float* __restrict__ feat,
    const _Float16* __restrict__ wihT, const _Float16* __restrict__ whhT,
    const float* __restrict__ w_ih0,
    const float* __restrict__ b_ih, const float* __restrict__ b_hh,
    _Float16* __restrict__ hseq, unsigned* __restrict__ bar) {
  const int wg   = blockIdx.x;
  const int tid  = threadIdx.x;
  const int wave = tid >> 5;
  const int lane = tid & 31;
  const int g    = wave >> 2;        // gate 0..3 (i,f,g,o)
  const int kq   = wave & 3;         // K quarter 0..3
  const int K0   = kq * 128;
  const int rowbase = g * Hn + wg * HS;  // 16 contiguous weight rows

  __shared__ float gatesLds[4][Bn][NRr + 4];   // per-K-quarter partials
  __shared__ float cLds[Bn][HS];
  __shared__ float biasLds[NRr];
  __shared__ float wih0Lds[NRr][8];

  // unique weight B-fragments, register resident (8 frags = 64 VGPRs)
  v16h bhh[4];
  v16h bih[4];
#pragma unroll
  for (int kk = 0; kk < 4; ++kk) {
    bhh[kk] = load_b_frag16(whhT, rowbase, K0 + kk * 32, 2048);
    if (!LAYER0) bih[kk] = load_b_frag16(wihT, rowbase, K0 + kk * 32, 2048);
  }

  for (int r = tid; r < NRr; r += 512) {
    int gg = r >> 4, j = r & 15;
    int grow = gg * Hn + wg * HS + j;
    biasLds[r] = b_ih[grow] + b_hh[grow];
    if (LAYER0) {
#pragma unroll
      for (int d = 0; d < 6; ++d) wih0Lds[r][d] = w_ih0[grow * 6 + d];
    }
  }
  for (int i = tid; i < Bn * HS; i += 512) cLds[i / HS][i % HS] = 0.0f;
  __syncthreads();

  for (int t = 0; t < Tn; ++t) {
    const size_t hbase = (size_t)t * (Bn * LDK) + K0;
    const size_t xbase = (size_t)(t + 1) * (Bn * LDK) + K0;
#pragma unroll
    for (int m = 0; m < 4; ++m) {
      v8f acc = {};
      const _Float16* hA = hseq + hbase + (size_t)(m * 16) * LDK;
#pragma unroll
      for (int kk = 0; kk < 4; ++kk) {
        v16h ah = load_a_frag(hA + kk * 32);
        acc = WMMA_F16(ah, bhh[kk], acc);
        if (!LAYER0) {
          const _Float16* xA = xin + xbase + (size_t)(m * 16) * LDK;
          v16h ax = load_a_frag(xA + kk * 32);
          acc = WMMA_F16(ax, bih[kk], acc);
        }
      }
      int n = lane & 15, hi = lane >> 4;
#pragma unroll
      for (int r = 0; r < 8; ++r)
        gatesLds[kq][m * 16 + r + hi * 8][g * 16 + n] = acc[r];
    }
    __syncthreads();

    // activations: 64*16 = 1024 (b,j) cells, 2 per thread; sum K-partials
    for (int idx = tid; idx < Bn * HS; idx += 512) {
      int bb = idx / HS, j = idx % HS;
      float gi = biasLds[j];
      float gf = biasLds[16 + j];
      float gg = biasLds[32 + j];
      float go = biasLds[48 + j];
#pragma unroll
      for (int q = 0; q < 4; ++q) {
        gi += gatesLds[q][bb][j];
        gf += gatesLds[q][bb][16 + j];
        gg += gatesLds[q][bb][32 + j];
        go += gatesLds[q][bb][48 + j];
      }
      if (LAYER0) {
#pragma unroll
        for (int d = 0; d < 6; ++d) {
          float xv = feat[(size_t)(bb * 6 + d) * Tn + t];
          gi += wih0Lds[j][d] * xv;
          gf += wih0Lds[16 + j][d] * xv;
          gg += wih0Lds[32 + j][d] * xv;
          go += wih0Lds[48 + j][d] * xv;
        }
      }
      float iv = sigmoidf_(gi);
      float fv = sigmoidf_(gf);
      float gv = tanhf(gg);
      float ov = sigmoidf_(go);
      float cv = fv * cLds[bb][j] + iv * gv;
      cLds[bb][j] = cv;
      float hv = ov * tanhf(cv);
      hseq[(size_t)(t + 1) * (Bn * LDK) + (size_t)bb * LDK + wg * HS + j] =
          (_Float16)hv;
    }
    grid_barrier(bar, (unsigned)(t + 1) * NWG);
  }
}

// ---------------------------------------------------------------------------
// Output projection: y(128*64 x 512) @ w_out^T + b_out. One 16x16 tile/wave.
// ---------------------------------------------------------------------------
__global__ __launch_bounds__(256) void proj_kernel(
    const _Float16* __restrict__ h2, const _Float16* __restrict__ woutT,
    const float* __restrict__ b_out, float* __restrict__ out) {
  int wave = threadIdx.x >> 5;
  int lane = threadIdx.x & 31;
  int tile = blockIdx.x * 8 + wave;
  int mt = tile >> 4;   // 0..511
  int nt = tile & 15;   // 0..15
  int r = lane & 15, hi = lane >> 4;
  int row = mt * 16 + r;
  int tt = row >> 6, bb = row & 63;
  const _Float16* arow =
      h2 + (size_t)(8 * tt + 8) * (Bn * LDK) + (size_t)bb * LDK + hi * 8;
  v8f acc = {};
#pragma unroll
  for (int kk = 0; kk < 16; ++kk) {
    v8h lo = *(const v8h*)(arow + kk * 32);
    v8h hh = *(const v8h*)(arow + kk * 32 + 16);
    v16h a;
#pragma unroll
    for (int i = 0; i < 8; ++i) { a[i] = lo[i]; a[8 + i] = hh[i]; }
    v16h b = load_b_frag16(woutT, nt * 16, kk * 32, 256);
    acc = WMMA_F16(a, b, acc);
  }
  int n = lane & 15;
  float bias = b_out[nt * 16 + n];
#pragma unroll
  for (int rr = 0; rr < 8; ++rr) {
    int orow = mt * 16 + rr + hi * 8;
    out[(size_t)orow * 256 + nt * 16 + n] = acc[rr] + bias;
  }
}

// one-time weight swizzle: W (R x 512 f32, row-major) -> WT (512 x R f16)
__global__ void prep_weights(const float* __restrict__ W,
                             _Float16* __restrict__ WT, int R) {
  size_t total = (size_t)R * 512;
  for (size_t i = (size_t)blockIdx.x * blockDim.x + threadIdx.x; i < total;
       i += (size_t)gridDim.x * blockDim.x) {
    int r = (int)(i / 512);
    int k = (int)(i % 512);
    WT[(size_t)k * R + r] = (_Float16)W[i];
  }
}

// init: zero barrier counters + the h_{-1} slice of both sequence buffers
__global__ void init_kernel(unsigned* __restrict__ bar,
                            _Float16* __restrict__ bufA,
                            _Float16* __restrict__ bufB) {
  size_t i = (size_t)blockIdx.x * blockDim.x + threadIdx.x;
  if (i < 64) bar[i] = 0u;
  if (i < (size_t)Bn * LDK) {
    bufA[i] = (_Float16)0.0f;
    bufB[i] = (_Float16)0.0f;
  }
}

extern "C" void kernel_launch(void* const* d_in, const int* in_sizes, int n_in,
                              void* d_out, int out_size, void* d_ws,
                              size_t ws_size, hipStream_t stream) {
  const float* feat  = (const float*)d_in[0];
  const float* w_ih0 = (const float*)d_in[1];
  const float* w_hh0 = (const float*)d_in[2];
  const float* b_ih0 = (const float*)d_in[3];
  const float* b_hh0 = (const float*)d_in[4];
  const float* w_ih1 = (const float*)d_in[5];
  const float* w_hh1 = (const float*)d_in[6];
  const float* b_ih1 = (const float*)d_in[7];
  const float* b_hh1 = (const float*)d_in[8];
  const float* w_ih2 = (const float*)d_in[9];
  const float* w_hh2 = (const float*)d_in[10];
  const float* b_ih2 = (const float*)d_in[11];
  const float* b_hh2 = (const float*)d_in[12];
  const float* w_out = (const float*)d_in[13];
  const float* b_out = (const float*)d_in[14];

  char* ws = (char*)d_ws;
  unsigned* bar = (unsigned*)ws;               // counters at +0, +16, +32
  _Float16* bufA = (_Float16*)(ws + 256);
  size_t seq_elems = (size_t)(Tn + 1) * Bn * LDK;
  _Float16* bufB  = bufA + seq_elems;
  _Float16* whh0T = bufB + seq_elems;
  size_t wt_elems = (size_t)512 * 2048;
  _Float16* wih1T = whh0T + wt_elems;
  _Float16* whh1T = wih1T + wt_elems;
  _Float16* wih2T = whh1T + wt_elems;
  _Float16* whh2T = wih2T + wt_elems;
  _Float16* woutT = whh2T + wt_elems;          // 512 x 256

  init_kernel<<<128, 256, 0, stream>>>(bar, bufA, bufB);
  prep_weights<<<1024, 256, 0, stream>>>(w_hh0, whh0T, 2048);
  prep_weights<<<1024, 256, 0, stream>>>(w_ih1, wih1T, 2048);
  prep_weights<<<1024, 256, 0, stream>>>(w_hh1, whh1T, 2048);
  prep_weights<<<1024, 256, 0, stream>>>(w_ih2, wih2T, 2048);
  prep_weights<<<1024, 256, 0, stream>>>(w_hh2, whh2T, 2048);
  prep_weights<<<256, 256, 0, stream>>>(w_out, woutT, 256);

  lstm_layer<true><<<NWG, 512, 0, stream>>>(bufB, feat, whh0T, whh0T, w_ih0,
                                            b_ih0, b_hh0, bufA, bar + 0);
  lstm_layer<false><<<NWG, 512, 0, stream>>>(bufA, feat, wih1T, whh1T, w_ih0,
                                             b_ih1, b_hh1, bufB, bar + 16);
  lstm_layer<false><<<NWG, 512, 0, stream>>>(bufB, feat, wih2T, whh2T, w_ih0,
                                             b_ih2, b_hh2, bufA, bar + 32);

  proj_kernel<<<1024, 256, 0, stream>>>(bufA, woutT, b_out, (float*)d_out);
}
